// Attention_52922587021751
// MI455X (gfx1250) — matrix operand order repaired and checked by reading, not compiled, not run on hardware
//
#include <hip/hip_runtime.h>
#include <math.h>

#define Bb 64
#define Ss 2048
#define Hh 1024
#define NEGV (-1.0e10f)

typedef float  v8f   __attribute__((ext_vector_type(8)));
typedef __bf16 v16bf __attribute__((ext_vector_type(16)));

union FragU { uint4 u[2]; v16bf v; };

__device__ __forceinline__ unsigned int pack_bf16x2(float lo, float hi) {
    unsigned int ulo = __float_as_uint(lo);
    unsigned int uhi = __float_as_uint(hi);
    ulo += 0x7FFFu + ((ulo >> 16) & 1u);   // round-to-nearest-even bf16
    uhi += 0x7FFFu + ((uhi >> 16) & 1u);
    return (ulo >> 16) | (uhi & 0xFFFF0000u);
}

// Branchless tanh: v_exp_f32 (trans) + fast divide; avoids libm's exec-mask path.
__device__ __forceinline__ float fast_tanh(float x) {
    x = fminf(fmaxf(x, -15.0f), 15.0f);
    float e = __expf(2.0f * x);
    return __fdividef(e - 1.0f, e + 1.0f);
}

// ---------------------------------------------------------------------------
// Kernel 1: dbias[b,k] = decoder_hidden[b,:] @ Wd[:,k] + b_attn[k]
// ---------------------------------------------------------------------------
__global__ __launch_bounds__(256) void k_decproj(const float* __restrict__ dh,
                                                 const float* __restrict__ W,
                                                 const float* __restrict__ ba,
                                                 float* __restrict__ dbias) {
    int idx = blockIdx.x * 256 + threadIdx.x;   // b*H + k
    int b = idx >> 10;
    int k = idx & (Hh - 1);
    const float* d = dh + b * Hh;
    float s = ba[k];
#pragma unroll 4
    for (int h = 0; h < Hh; ++h)
        s = fmaf(d[h], W[(size_t)h * Hh + k], s);   // coalesced over k
    dbias[idx] = s;
}

// ---------------------------------------------------------------------------
// Kernel 2: pack We = W_attn[H:2H, :] into bf16 WMMA B-fragments
// (ISA 7.12.2, 16-bit B 32x16), lane-contiguous 32B per lane per fragment.
// ---------------------------------------------------------------------------
__global__ __launch_bounds__(256) void k_pack(const float* __restrict__ W,
                                              unsigned int* __restrict__ wp) {
    int idx  = blockIdx.x * 256 + threadIdx.x;   // one dword (2 bf16) each
    int v    = idx & 7;                          // VGPR index within fragment
    int lane = (idx >> 3) & 31;
    int frag = idx >> 8;
    int n_idx = frag & 63;
    int kc    = frag >> 6;
    int col = n_idx * 16 + (lane & 15);
    int k0  = kc * 32 + ((lane & 16) ? 8 : 0) + ((v & 4) << 2) + 2 * (v & 3);
    const float* We = W + (size_t)Hh * Hh;       // second half of W_attn
    float lo = We[(size_t)k0 * Hh + col];
    float hi = We[(size_t)(k0 + 1) * Hh + col];
    wp[idx] = pack_bf16x2(lo, hi);
}

// ---------------------------------------------------------------------------
// Kernel 3: fused energy kernel. One workgroup = 32 encoder rows of one batch
// (two 16-row M-blocks -> 2x B-fragment reuse). 8 waves x 128 N-columns.
// Per K-step: stage 32x32 A tile to LDS (bf16, double-buffered, software-
// pipelined E loads), batch-load 8 B fragments, issue 16 WMMAs back-to-back.
// Epilogue fuses tanh + v-dot + mask.
// ---------------------------------------------------------------------------
__global__ __launch_bounds__(256) void k_energy(const float* __restrict__ E,
                                                const float* __restrict__ dbias,
                                                const unsigned int* __restrict__ wp,
                                                const float* __restrict__ vw,
                                                const int* __restrict__ mask,
                                                float* __restrict__ att) {
    __shared__ unsigned int ldsA[2][512];   // 32 rows x 32 bf16 (16 dwords/row)
    __shared__ float red[32][8];

    const int b    = blockIdx.y;
    const int s0   = blockIdx.x * 32;
    const int tid  = threadIdx.x;
    const int wave = tid >> 5;
    const int lane = tid & 31;
    const int l15  = lane & 15;
    const int half = lane >> 4;

    v8f zero = {};
    v8f acc0[8], acc1[8];
#pragma unroll
    for (int i = 0; i < 8; ++i) { acc0[i] = zero; acc1[i] = zero; }

    // A-tile staging: thread -> (row ar 0..31, k-quad aq 0..7); 1024 floats/step.
    const int ar = tid >> 3;
    const int aq = tid & 7;
    const float4* Ep =
        reinterpret_cast<const float4*>(E + ((size_t)b * Ss + s0 + ar) * Hh) + aq;
    const int ldsWr = ar * 16 + aq * 2;      // dword index in ldsA buffer

    // A-fragment LDS dword offsets (ISA 7.12.2, 16-bit A 16x32):
    // lanes 0-15: K 0..7 then 16..23 ; lanes 16-31: K 8..15 then 24..31.
    const unsigned aOff0 = (unsigned)((l15 << 4) + (half << 2));          // rows 0..15
    const unsigned aOff1 = (unsigned)(((16 + l15) << 4) + (half << 2));   // rows 16..31
    const unsigned int* wpl = wp + (lane << 3);

    float4 e = Ep[0];                        // software-pipelined E load
    for (int kc = 0; kc < 32; ++kc) {
        // Stage the tile loaded last iteration into this step's LDS buffer.
        uint2 pk;
        pk.x = pack_bf16x2(e.x, e.y);
        pk.y = pack_bf16x2(e.z, e.w);
        *reinterpret_cast<uint2*>(&ldsA[kc & 1][ldsWr]) = pk;
        __syncthreads();

        // Batch-load all 8 B fragments (L2-resident packed We).
        FragU bu[8];
#pragma unroll
        for (int nc = 0; nc < 8; ++nc) {
            const uint4* pb = reinterpret_cast<const uint4*>(
                wpl + (((size_t)kc * 64 + wave * 8 + nc) << 8));
            bu[nc].u[0] = pb[0];
            bu[nc].u[1] = pb[1];
        }

        // Two A fragments (M-blocks 0 and 1).
        FragU a0, a1;
        const uint4* pa0 = reinterpret_cast<const uint4*>(&ldsA[kc & 1][aOff0]);
        a0.u[0] = pa0[0];
        a0.u[1] = pa0[2];
        const uint4* pa1 = reinterpret_cast<const uint4*>(&ldsA[kc & 1][aOff1]);
        a1.u[0] = pa1[0];
        a1.u[1] = pa1[2];

        // Kick off next step's E load; overlaps the WMMA burst below.
        int kn = (kc < 31) ? kc + 1 : 31;
        e = Ep[(size_t)kn * 8];

#pragma unroll
        for (int nc = 0; nc < 8; ++nc) {
            acc0[nc] = __builtin_amdgcn_wmma_f32_16x16x32_bf16(
                false, a0.v, false, bu[nc].v, (short)0, acc0[nc], false, false);
            acc1[nc] = __builtin_amdgcn_wmma_f32_16x16x32_bf16(
                false, a1.v, false, bu[nc].v, (short)0, acc1[nc], false, false);
        }
    }

    // Epilogue: C layout (f32 16x16): column = l15, rows half*8 + j in VGPR j.
    float p0[8], p1[8];
#pragma unroll
    for (int j = 0; j < 8; ++j) { p0[j] = 0.f; p1[j] = 0.f; }
#pragma unroll
    for (int nc = 0; nc < 8; ++nc) {
        int n = wave * 128 + nc * 16 + l15;
        float wgt = vw[n];
        float db  = dbias[b * Hh + n];
#pragma unroll
        for (int j = 0; j < 8; ++j) {
            p0[j] += wgt * fast_tanh(acc0[nc][j] + db);
            p1[j] += wgt * fast_tanh(acc1[nc][j] + db);
        }
    }
    // Reduce across the 16 columns held within each half-wave.
#pragma unroll
    for (int j = 0; j < 8; ++j) {
#pragma unroll
        for (int d = 1; d <= 8; d <<= 1) {
            p0[j] += __shfl_xor(p0[j], d, 32);
            p1[j] += __shfl_xor(p1[j], d, 32);
        }
    }
    if (l15 == 0) {
#pragma unroll
        for (int j = 0; j < 8; ++j) {
            red[half * 8 + j][wave]      = p0[j];
            red[16 + half * 8 + j][wave] = p1[j];
        }
    }
    __syncthreads();
    if (tid < 32) {
        float s = 0.f;
#pragma unroll
        for (int w = 0; w < 8; ++w) s += red[tid][w];
        int sidx = s0 + tid;
        if (mask[(size_t)b * Ss + sidx] == 1) s = NEGV;
        att[(size_t)b * Ss + sidx] = s;
    }
}

// ---------------------------------------------------------------------------
// Kernel 4: row softmax over S=2048. One block per batch row.
// ---------------------------------------------------------------------------
__global__ __launch_bounds__(256) void k_softmax(const float* __restrict__ att,
                                                 float* __restrict__ out) {
    __shared__ float smax[8];
    __shared__ float ssum[8];
    const int b = blockIdx.x;
    const int tid = threadIdx.x, wave = tid >> 5, lane = tid & 31;
    const float* row = att + (size_t)b * Ss;

    float vals[8];
    float m = -3.4e38f;
#pragma unroll
    for (int i = 0; i < 8; ++i) {
        vals[i] = row[tid + i * 256];
        m = fmaxf(m, vals[i]);
    }
#pragma unroll
    for (int d = 16; d >= 1; d >>= 1) m = fmaxf(m, __shfl_xor(m, d, 32));
    if (lane == 0) smax[wave] = m;
    __syncthreads();
    float bm = smax[0];
#pragma unroll
    for (int w = 1; w < 8; ++w) bm = fmaxf(bm, smax[w]);

    float sum = 0.f;
#pragma unroll
    for (int i = 0; i < 8; ++i) {
        float e = __expf(vals[i] - bm);
        vals[i] = e;
        sum += e;
    }
#pragma unroll
    for (int d = 16; d >= 1; d >>= 1) sum += __shfl_xor(sum, d, 32);
    if (lane == 0) ssum[wave] = sum;
    __syncthreads();
    float tot = 0.f;
#pragma unroll
    for (int w = 0; w < 8; ++w) tot += ssum[w];
    float inv = 1.0f / tot;
#pragma unroll
    for (int i = 0; i < 8; ++i) out[(size_t)b * Ss + tid + i * 256] = vals[i] * inv;
}

// ---------------------------------------------------------------------------
extern "C" void kernel_launch(void* const* d_in, const int* in_sizes, int n_in,
                              void* d_out, int out_size, void* d_ws, size_t ws_size,
                              hipStream_t stream) {
    (void)in_sizes; (void)n_in; (void)out_size; (void)ws_size;
    const float* dh   = (const float*)d_in[0];   // decoder_hidden (B,H)
    const float* E    = (const float*)d_in[1];   // encoder_outputs (B,S,H)
    const int*   msk  = (const int*)d_in[2];     // mask (B,S)
    const float* W    = (const float*)d_in[3];   // W_attn (2H,H)
    const float* ba   = (const float*)d_in[4];   // b_attn (H)
    const float* vwp  = (const float*)d_in[5];   // v_w (H)
    float* out = (float*)d_out;                  // (B,S)

    char* ws = (char*)d_ws;
    float*        dbias = (float*)ws;                                   // 256 KB
    unsigned int* wpack = (unsigned int*)(ws + (size_t)Bb * Hh * 4);    // 2 MB
    float*        att   = (float*)(ws + (size_t)Bb * Hh * 4 + (size_t)Hh * Hh * 2);

    k_decproj<<<(Bb * Hh) / 256, 256, 0, stream>>>(dh, W, ba, dbias);
    k_pack<<<(Hh * Hh / 2) / 256, 256, 0, stream>>>(W, wpack);
    dim3 grid(Ss / 32, Bb);
    k_energy<<<grid, 256, 0, stream>>>(E, dbias, wpack, vwp, msk, att);
    k_softmax<<<Bb, 256, 0, stream>>>(att, out);
}
